// CCAttention_49632642073374
// MI455X (gfx1250) — compile-verified, matrix operand order checked by reading, not run on hardware
//
#include <hip/hip_runtime.h>
#include <hip/hip_bf16.h>
#include <math.h>

typedef _Float16 h16;
typedef __attribute__((ext_vector_type(16))) _Float16 v16h;
typedef __attribute__((ext_vector_type(4)))  _Float16 v4h;
typedef __attribute__((ext_vector_type(8)))  float    v8f;

#ifndef INF_NEG
#define INF_NEG (-1e30f)
#endif

union FragU { v16h v; float4 q[2]; };

// Load a 16x32 f16 WMMA operand fragment (A-layout; B for "row-major K^T" uses
// the identical pattern with row = column index).
// ISA layout (16-bit A 16x32): lanes 0-15: row=lane, K chunks {0..7,16..23};
// lanes 16-31: row=lane-16, K chunks {8..15,24..31}.
__device__ __forceinline__ v16h load_frag16(const h16* __restrict__ base, int stride, int lane) {
  int hs = lane >> 4;
  int r  = lane & 15;
  const h16* p = base + (size_t)r * stride + hs * 8;
  FragU f;
  f.q[0] = *(const float4*)(p);
  f.q[1] = *(const float4*)(p + 16);
  return f.v;
}

// ---------------------------------------------------------------------------
// Generic WMMA GEMM:  C[m,n] = sum_k A[m,k] * W[n,k]  (+ bias[n])
// A: MxK f16 row-major (lda), W: NxK f16 row-major (ldw), C: MxN f32 (ldc).
// M mult of 16; N mult of 32; K mult of 32. One wave -> one 16x32 tile
// (A fragment reused across two WMMAs), software-pipelined fragment loads.
// ---------------------------------------------------------------------------
__global__ __launch_bounds__(128) void gemm_wmma_f16(
    const h16* __restrict__ A, int lda,
    const h16* __restrict__ W, int ldw,
    float* __restrict__ Cmat, int ldc,
    int M, int N, int K, const float* __restrict__ bias)
{
  int lane = threadIdx.x & 31;
  int wave = threadIdx.x >> 5;
  int tilesN = N >> 5;                    // 32-wide N tiles
  int tile = blockIdx.x * 4 + wave;
  int tm = tile / tilesN;
  int tn = tile - tm * tilesN;
  if (tm * 16 >= M) return;
  const h16* Ab  = A + (size_t)tm * 16 * lda;
  const h16* Wb0 = W + (size_t)(tn * 32) * ldw;
  const h16* Wb1 = Wb0 + (size_t)16 * ldw;

  v8f acc0 = {}, acc1 = {};
  // prologue fragments
  v16h af  = load_frag16(Ab,  lda, lane);
  v16h bf0 = load_frag16(Wb0, ldw, lane);
  v16h bf1 = load_frag16(Wb1, ldw, lane);
  for (int k0 = 0;;) {
    int kn = k0 + 32;
    v16h afn, bf0n, bf1n;
    if (kn < K) {
      // issue next-iteration loads BEFORE consuming current fragments so the
      // WMMAs below overlap with the in-flight vmem
      __builtin_prefetch(Ab  + (size_t)(lane & 15) * lda + kn + 32, 0, 3);
      __builtin_prefetch(Wb0 + (size_t)(lane & 15) * ldw + kn + 32, 0, 3);
      afn  = load_frag16(Ab  + kn, lda, lane);
      bf0n = load_frag16(Wb0 + kn, ldw, lane);
      bf1n = load_frag16(Wb1 + kn, ldw, lane);
    }
    acc0 = __builtin_amdgcn_wmma_f32_16x16x32_f16(false, af, false, bf0,
                                                  (short)0, acc0, false, false);
    acc1 = __builtin_amdgcn_wmma_f32_16x16x32_f16(false, af, false, bf1,
                                                  (short)0, acc1, false, false);
    if (kn >= K) break;
    af = afn; bf0 = bf0n; bf1 = bf1n;
    k0 = kn;
  }
  int n0 = tn * 32 + (lane & 15);
  float bv0 = bias ? bias[n0]      : 0.0f;
  float bv1 = bias ? bias[n0 + 16] : 0.0f;
  int rbase = tm * 16 + (lane >> 4) * 8;
#pragma unroll
  for (int r = 0; r < 8; ++r) {
    Cmat[(size_t)(rbase + r) * ldc + n0]      = acc0[r] + bv0;
    Cmat[(size_t)(rbase + r) * ldc + n0 + 16] = acc1[r] + bv1;
  }
}

// ---------------------------------------------------------------------------
// Flash-style attention, one wave per 16-row Q tile, 32-key chunks staged in
// double-buffered LDS (stage of chunk k+1 overlaps compute of chunk k).
// Q: (L x ldq) f16, head cols at h*hd. K/V: (Lk x ldkv) f16, head cols h*hd.
// O: f16 (L x ldo), written at col ocol0 + h*hd. hd in {16,32} (padded to 32).
// ---------------------------------------------------------------------------
#define ATT_WAVES 8
__global__ __launch_bounds__(256) void attn_wmma(
    const h16* __restrict__ Q, int ldq,
    const h16* __restrict__ Km, const h16* __restrict__ Vm, int ldkv,
    h16* __restrict__ O, int ldo, int ocol0,
    int Lk, int hd, float scale)
{
  __shared__ __align__(16) h16 sK [2][32 * 32];
  __shared__ __align__(16) h16 sVt[2][32 * 32];
  __shared__ __align__(16) h16 sP [ATT_WAVES * 16 * 32];

  int tid  = threadIdx.x;
  int lane = tid & 31;
  int wave = tid >> 5;
  int h    = blockIdx.y;
  int qt   = blockIdx.x * ATT_WAVES + wave;
  int qrow0 = qt * 16;
  int qcol  = h * hd;

  // Q fragment, kept in registers for the whole key loop (zero-padded K-dim)
  v16h qf;
  {
    int hs = lane >> 4, r = lane & 15;
    const h16* p = Q + (size_t)(qrow0 + r) * ldq + qcol + hs * 8;
    FragU f;
    f.q[0] = *(const float4*)p;
    if (hd > 16) f.q[1] = *(const float4*)(p + 16);
    else         f.q[1] = make_float4(0.f, 0.f, 0.f, 0.f);
    qf = f.v;
  }

  // cooperative stage: K row-major, V transposed; zero pad keys>=Lk, dims>=hd
  auto stage = [&](int ck, int buf) {
    int kbase = ck << 5;
    for (int i = tid; i < 32 * 32; i += 256) {
      int kr = i >> 5, c = i & 31;
      int key = kbase + kr;
      bool ok = (key < Lk) && (c < hd);
      h16 kk = ok ? Km[(size_t)key * ldkv + qcol + c] : (h16)0.f;
      h16 vv = ok ? Vm[(size_t)key * ldkv + qcol + c] : (h16)0.f;
      sK[buf][i] = kk;
      sVt[buf][c * 32 + kr] = vv;
    }
  };

  h16* sPw = sP + wave * (16 * 32);
  float mrun[8], lrun[8];
  v8f o0 = {}, o1 = {};
#pragma unroll
  for (int r = 0; r < 8; ++r) { mrun[r] = INF_NEG; lrun[r] = 0.f; }

  int nchunk = (Lk + 31) >> 5;
  stage(0, 0);
  __syncthreads();

  for (int ck = 0; ck < nchunk; ++ck) {
    int buf = ck & 1;
    if (ck + 1 < nchunk) stage(ck + 1, buf ^ 1);   // overlaps compute below

    int kbase = ck << 5;
    v16h kb0 = load_frag16(sK[buf], 32, lane);
    v16h kb1 = load_frag16(sK[buf] + 16 * 32, 32, lane);
    v8f zf = {};
    v8f s0 = __builtin_amdgcn_wmma_f32_16x16x32_f16(false, qf, false, kb0,
                                                    (short)0, zf, false, false);
    v8f s1 = __builtin_amdgcn_wmma_f32_16x16x32_f16(false, qf, false, kb1,
                                                    (short)0, zf, false, false);

    int n15 = lane & 15;
    bool ok0 = (kbase + n15) < Lk;
    bool ok1 = (kbase + 16 + n15) < Lk;
    int prow0 = (lane >> 4) * 8;
#pragma unroll
    for (int r = 0; r < 8; ++r) {
      float a = ok0 ? s0[r] * scale : INF_NEG;
      float b = ok1 ? s1[r] * scale : INF_NEG;
      float mx = fmaxf(a, b);
      mx = fmaxf(mx, __shfl_xor(mx, 1));
      mx = fmaxf(mx, __shfl_xor(mx, 2));
      mx = fmaxf(mx, __shfl_xor(mx, 4));
      mx = fmaxf(mx, __shfl_xor(mx, 8));
      float mnew  = fmaxf(mrun[r], mx);
      float alpha = __expf(mrun[r] - mnew);
      float p0 = ok0 ? __expf(a - mnew) : 0.f;
      float p1 = ok1 ? __expf(b - mnew) : 0.f;
      float rs = p0 + p1;
      rs += __shfl_xor(rs, 1);
      rs += __shfl_xor(rs, 2);
      rs += __shfl_xor(rs, 4);
      rs += __shfl_xor(rs, 8);
      lrun[r] = lrun[r] * alpha + rs;
      mrun[r] = mnew;
      o0[r] *= alpha;
      o1[r] *= alpha;
      sPw[(prow0 + r) * 32 + n15]      = (h16)p0;
      sPw[(prow0 + r) * 32 + 16 + n15] = (h16)p1;
    }
    // make per-wave LDS P writes visible before ds_load fragments
    asm volatile("s_wait_dscnt 0" ::: "memory");

    v16h pf  = load_frag16(sPw, 32, lane);
    v16h vb0 = load_frag16(sVt[buf], 32, lane);
    o0 = __builtin_amdgcn_wmma_f32_16x16x32_f16(false, pf, false, vb0,
                                                (short)0, o0, false, false);
    if (hd > 16) {
      v16h vb1 = load_frag16(sVt[buf] + 16 * 32, 32, lane);
      o1 = __builtin_amdgcn_wmma_f32_16x16x32_f16(false, pf, false, vb1,
                                                  (short)0, o1, false, false);
    }
    __syncthreads();   // staging of ck+1 done; safe to flip buffers
  }

  int col = ocol0 + qcol + (lane & 15);
#pragma unroll
  for (int r = 0; r < 8; ++r) {
    int row = qrow0 + (lane >> 4) * 8 + r;
    float inv = 1.0f / lrun[r];
    O[(size_t)row * ldo + col] = (h16)(o0[r] * inv);
    if (hd > 16) O[(size_t)row * ldo + col + 16] = (h16)(o1[r] * inv);
  }
}

// ---------------------------------------------------------------------------
// Row-wise LayerNorm (+ optional exact GELU), f32 in -> f16 out. Wave per row.
// ---------------------------------------------------------------------------
__global__ __launch_bounds__(128) void ln_rows(
    const float* __restrict__ X, int D,
    const float* __restrict__ w, const float* __restrict__ b,
    h16* __restrict__ out16, int do_gelu, int rows)
{
  int wave = threadIdx.x >> 5, lane = threadIdx.x & 31;
  int row = blockIdx.x * 4 + wave;
  if (row >= rows) return;
  const float* x = X + (size_t)row * D;
  float s = 0.f, ss = 0.f;
  for (int i = lane; i < D; i += 32) { float v = x[i]; s += v; ss += v * v; }
  for (int m = 16; m; m >>= 1) { s += __shfl_xor(s, m); ss += __shfl_xor(ss, m); }
  float mean = s / D;
  float var  = ss / D - mean * mean;
  float rstd = rsqrtf(var + 1e-5f);
  for (int i = lane; i < D; i += 32) {
    float v = (x[i] - mean) * rstd * w[i] + b[i];
    if (do_gelu) v = 0.5f * v * (1.f + erff(v * 0.70710678118654752f));
    out16[(size_t)row * D + i] = (h16)v;
  }
}

// f32 -> f16, 4 elements per thread (n must be a multiple of 4)
__global__ void cvt_f32_f16_v4(const float4* __restrict__ s, v4h* __restrict__ d, int n4) {
  int i = blockIdx.x * blockDim.x + threadIdx.x;
  if (i >= n4) return;
  float4 t = s[i];
  v4h o;
  o[0] = (h16)t.x; o[1] = (h16)t.y; o[2] = (h16)t.z; o[3] = (h16)t.w;
  d[i] = o;
}

// Patchify (stride==kernel conv -> GEMM): P[p][(ky*s+kx)*C + c]
__global__ void patchify(const h16* __restrict__ xn, int Cin, int Wimg, int s, int Wo,
                         h16* __restrict__ P, int total) {
  int idx = blockIdx.x * blockDim.x + threadIdx.x;
  if (idx >= total) return;
  int Kp  = s * s * Cin;
  int p   = idx / Kp, r = idx - p * Kp;
  int pix = r / Cin,  c = r - pix * Cin;
  int ky = pix / s,   kx = pix - ky * s;
  int py = p / Wo,    px = p - py * Wo;
  P[idx] = xn[((size_t)((py * s + ky) * Wimg + (px * s + kx))) * Cin + c];
}

// w[o][c][ky][kx] (f32) -> wt[o][(ky*s+kx)*Cin + c] (f16)
__global__ void reorder_w(const float* __restrict__ w, int Cin, int s,
                          h16* __restrict__ wt, int total) {
  int idx = blockIdx.x * blockDim.x + threadIdx.x;
  if (idx >= total) return;
  int Kp  = Cin * s * s;
  int o   = idx / Kp, r = idx - o * Kp;
  int pix = r / Cin,  c = r - pix * Cin;
  int ky = pix / s,   kx = pix - ky * s;
  wt[idx] = (h16)w[((size_t)(o * Cin + c) * s + ky) * s + kx];
}

// Depthwise 3x3 conv (pad 1) over v image (channel ce at column vcol0+ce of kv),
// output n-major (N x Ce) with bias included.
__global__ void dwconv(const float* __restrict__ kv, int ldkv, int vcol0,
                       int Ce, int Hs, int Ws,
                       const float* __restrict__ w, const float* __restrict__ b,
                       float* __restrict__ dout) {
  int idx = blockIdx.x * blockDim.x + threadIdx.x;
  int Nn = Hs * Ws;
  if (idx >= Nn * Ce) return;
  int n = idx / Ce, ce = idx - n * Ce;
  int y = n / Ws,   x = n - y * Ws;
  float acc = b[ce];
#pragma unroll
  for (int dy = -1; dy <= 1; ++dy)
#pragma unroll
    for (int dx = -1; dx <= 1; ++dx) {
      int yy = y + dy, xx = x + dx;
      if (yy >= 0 && yy < Hs && xx >= 0 && xx < Ws)
        acc += kv[(size_t)(yy * Ws + xx) * ldkv + vcol0 + ce] * w[ce * 9 + (dy + 1) * 3 + (dx + 1)];
    }
  dout[(size_t)n * Ce + ce] = acc;
}

// Replicate the reference's scrambled reshape-add:
//   v'_flat(ce-major)[t] = v_flat(ce-major)[t] + dconv_flat(n-major)[t]
// and also emit f16 copies of K.
__global__ void vk_finalize(const float* __restrict__ kv, int ldkv,
                            int kcol0, int vcol0, int Ce,
                            const float* __restrict__ dconv, int Nn,
                            h16* __restrict__ k16, h16* __restrict__ v16) {
  int idx = blockIdx.x * blockDim.x + threadIdx.x;
  if (idx >= Nn * Ce) return;
  int ce_t = idx / Nn, n2 = idx - ce_t * Nn;   // target, ce-major
  int n    = idx / Ce, ce = idx - n * Ce;      // source dconv, n-major
  float vv = kv[(size_t)n2 * ldkv + vcol0 + ce_t] + dconv[(size_t)n * Ce + ce];
  v16[(size_t)n2 * Ce + ce_t] = (h16)vv;
  k16[(size_t)n * Ce + ce]    = (h16)kv[(size_t)n * ldkv + kcol0 + ce];
}

// ---------------------------------------------------------------------------
extern "C" void kernel_launch(void* const* d_in, const int* in_sizes, int n_in,
                              void* d_out, int out_size, void* d_ws, size_t ws_size,
                              hipStream_t stream) {
  const int L = 9216, C = 256;
  const int N0 = 2304, N1 = 576, N2 = 144;        // reduced token counts
  const int K0 = 1024, K1 = 4096, K2 = 16384;     // patch GEMM depths
  const float SCALE = 0.17677669529663687f;       // (C/NH)^-0.5 = 1/sqrt(32)

  const float* x0        = (const float*)d_in[0];
  const float* x1        = (const float*)d_in[1];
  const float* msa_w     = (const float*)d_in[2];
  const float* msa_b     = (const float*)d_in[3];
  const float* red0_w    = (const float*)d_in[4];
  const float* red0_b    = (const float*)d_in[5];
  const float* red1_w    = (const float*)d_in[6];
  const float* red1_b    = (const float*)d_in[7];
  const float* red2_w    = (const float*)d_in[8];
  const float* red2_b    = (const float*)d_in[9];
  const float* q0_w      = (const float*)d_in[10];
  const float* q12_w     = (const float*)d_in[11];
  const float* kv0_w     = (const float*)d_in[12];
  const float* kv1_w     = (const float*)d_in[13];
  const float* kv2_w     = (const float*)d_in[14];
  const float* norm0_w   = (const float*)d_in[15];
  const float* norm0_b   = (const float*)d_in[16];
  const float* norm1_w   = (const float*)d_in[17];
  const float* norm1_b   = (const float*)d_in[18];
  const float* norm2_w   = (const float*)d_in[19];
  const float* norm2_b   = (const float*)d_in[20];
  const float* dwc0_w    = (const float*)d_in[21];
  const float* dwc0_b    = (const float*)d_in[22];
  const float* dwc1_w    = (const float*)d_in[23];
  const float* dwc1_b    = (const float*)d_in[24];
  const float* dwc2_w    = (const float*)d_in[25];
  const float* dwc2_b    = (const float*)d_in[26];
  const float* proj_w    = (const float*)d_in[27];
  const float* proj_b    = (const float*)d_in[28];
  float* out = (float*)d_out;

  // ---- workspace carve-out (total ~61 MB) ----
  char* ws = (char*)d_ws;
  size_t off = 0;
  auto alloc = [&](size_t bytes) -> void* {
    void* p = ws + off;
    off += (bytes + 255) & ~(size_t)255;
    return p;
  };
  h16*   xn16   = (h16*)alloc((size_t)L * C * 2);
  h16*   x016   = (h16*)alloc((size_t)L * C * 2);
  h16*   P0     = (h16*)alloc((size_t)N0 * K0 * 2);
  h16*   P1     = (h16*)alloc((size_t)N1 * K1 * 2);
  h16*   P2     = (h16*)alloc((size_t)N2 * K2 * 2);
  h16*   wt0    = (h16*)alloc((size_t)256 * K0 * 2);
  h16*   wt1    = (h16*)alloc((size_t)128 * K1 * 2);
  h16*   wt2    = (h16*)alloc((size_t)128 * K2 * 2);
  h16*   wq0    = (h16*)alloc((size_t)256 * 256 * 2);
  h16*   wq12   = (h16*)alloc((size_t)128 * 128 * 2);
  h16*   wkv0   = (h16*)alloc((size_t)256 * 256 * 2);
  h16*   wkv1   = (h16*)alloc((size_t)128 * 128 * 2);
  h16*   wkv2   = (h16*)alloc((size_t)128 * 128 * 2);
  h16*   wproj  = (h16*)alloc((size_t)256 * 256 * 2);
  float* qfull  = (float*)alloc((size_t)L * C * 4);
  h16*   qf16   = (h16*)alloc((size_t)L * C * 2);
  float* q12f   = (float*)alloc((size_t)L * 128 * 4);
  h16*   q12f16 = (h16*)alloc((size_t)L * 128 * 2);
  float* xs0    = (float*)alloc((size_t)N0 * 256 * 4);
  float* xs1    = (float*)alloc((size_t)N1 * 128 * 4);
  float* xs2    = (float*)alloc((size_t)N2 * 128 * 4);
  h16*   xg0    = (h16*)alloc((size_t)N0 * 256 * 2);
  h16*   xg1    = (h16*)alloc((size_t)N1 * 128 * 2);
  h16*   xg2    = (h16*)alloc((size_t)N2 * 128 * 2);
  float* kv0f   = (float*)alloc((size_t)N0 * 256 * 4);
  float* kv1f   = (float*)alloc((size_t)N1 * 128 * 4);
  float* kv2f   = (float*)alloc((size_t)N2 * 128 * 4);
  float* dc0    = (float*)alloc((size_t)N0 * 128 * 4);
  float* dc1    = (float*)alloc((size_t)N1 * 64 * 4);
  float* dc2    = (float*)alloc((size_t)N2 * 64 * 4);
  h16*   k016   = (h16*)alloc((size_t)N0 * 128 * 2);
  h16*   v016   = (h16*)alloc((size_t)N0 * 128 * 2);
  h16*   k116   = (h16*)alloc((size_t)N1 * 64 * 2);
  h16*   v116   = (h16*)alloc((size_t)N1 * 64 * 2);
  h16*   k216   = (h16*)alloc((size_t)N2 * 64 * 2);
  h16*   v216   = (h16*)alloc((size_t)N2 * 64 * 2);
  h16*   oc16   = (h16*)alloc((size_t)L * C * 2);
  (void)ws_size; (void)n_in; (void)in_sizes; (void)out_size;

  const int T = 256;
  auto nb = [](int n, int t) { return (n + t - 1) / t; };
  auto cvt = [&](const float* s, h16* d, int n) {
    cvt_f32_f16_v4<<<nb(n / 4, T), T, 0, stream>>>((const float4*)s, (v4h*)d, n / 4);
  };

  // ---- precision conversions ----
  cvt(x0, x016, L * C);
  cvt(q0_w, wq0, 256 * 256);
  cvt(q12_w, wq12, 128 * 128);
  cvt(kv0_w, wkv0, 256 * 256);
  cvt(kv1_w, wkv1, 128 * 128);
  cvt(kv2_w, wkv2, 128 * 128);
  cvt(proj_w, wproj, 256 * 256);

  // ---- msa LayerNorm, patchify, conv-weight reorder ----
  ln_rows<<<nb(L, 4), 128, 0, stream>>>(x1, C, msa_w, msa_b, xn16, 0, L);
  reorder_w<<<nb(256 * K0, T), T, 0, stream>>>(red0_w, 256, 2, wt0, 256 * K0);
  reorder_w<<<nb(128 * K1, T), T, 0, stream>>>(red1_w, 256, 4, wt1, 128 * K1);
  reorder_w<<<nb(128 * K2, T), T, 0, stream>>>(red2_w, 256, 8, wt2, 128 * K2);
  patchify<<<nb(N0 * K0, T), T, 0, stream>>>(xn16, 256, 96, 2, 48, P0, N0 * K0);
  patchify<<<nb(N1 * K1, T), T, 0, stream>>>(xn16, 256, 96, 4, 24, P1, N1 * K1);
  patchify<<<nb(N2 * K2, T), T, 0, stream>>>(xn16, 256, 96, 8, 12, P2, N2 * K2);

  // ---- q projections ----
  gemm_wmma_f16<<<nb((L / 16) * (256 / 32), 4), 128, 0, stream>>>(
      x016, 256, wq0, 256, qfull, 256, L, 256, 256, nullptr);
  cvt(qfull, qf16, L * C);
  gemm_wmma_f16<<<nb((L / 16) * (128 / 32), 4), 128, 0, stream>>>(
      qf16 + 128, 256, wq12, 128, q12f, 128, L, 128, 128, nullptr);
  cvt(q12f, q12f16, L * 128);

  // ---- reduction convs as GEMMs ----
  gemm_wmma_f16<<<nb((N0 / 16) * (256 / 32), 4), 128, 0, stream>>>(
      P0, K0, wt0, K0, xs0, 256, N0, 256, K0, red0_b);
  gemm_wmma_f16<<<nb((N1 / 16) * (128 / 32), 4), 128, 0, stream>>>(
      P1, K1, wt1, K1, xs1, 128, N1, 128, K1, red1_b);
  gemm_wmma_f16<<<nb((N2 / 16) * (128 / 32), 4), 128, 0, stream>>>(
      P2, K2, wt2, K2, xs2, 128, N2, 128, K2, red2_b);

  // ---- LN + GELU -> f16 ----
  ln_rows<<<nb(N0, 4), 128, 0, stream>>>(xs0, 256, norm0_w, norm0_b, xg0, 1, N0);
  ln_rows<<<nb(N1, 4), 128, 0, stream>>>(xs1, 128, norm1_w, norm1_b, xg1, 1, N1);
  ln_rows<<<nb(N2, 4), 128, 0, stream>>>(xs2, 128, norm2_w, norm2_b, xg2, 1, N2);

  // ---- kv projections ----
  gemm_wmma_f16<<<nb((N0 / 16) * (256 / 32), 4), 128, 0, stream>>>(
      xg0, 256, wkv0, 256, kv0f, 256, N0, 256, 256, nullptr);
  gemm_wmma_f16<<<nb((N1 / 16) * (128 / 32), 4), 128, 0, stream>>>(
      xg1, 128, wkv1, 128, kv1f, 128, N1, 128, 128, nullptr);
  gemm_wmma_f16<<<nb((N2 / 16) * (128 / 32), 4), 128, 0, stream>>>(
      xg2, 128, wkv2, 128, kv2f, 128, N2, 128, 128, nullptr);

  // ---- depthwise conv on v + scrambled add, emit f16 K/V ----
  dwconv<<<nb(N0 * 128, T), T, 0, stream>>>(kv0f, 256, 128, 128, 48, 48, dwc0_w, dwc0_b, dc0);
  dwconv<<<nb(N1 * 64,  T), T, 0, stream>>>(kv1f, 128, 64,  64,  24, 24, dwc1_w, dwc1_b, dc1);
  dwconv<<<nb(N2 * 64,  T), T, 0, stream>>>(kv2f, 128, 64,  64,  12, 12, dwc2_w, dwc2_b, dc2);
  vk_finalize<<<nb(N0 * 128, T), T, 0, stream>>>(kv0f, 256, 0, 128, 128, dc0, N0, k016, v016);
  vk_finalize<<<nb(N1 * 64,  T), T, 0, stream>>>(kv1f, 128, 0, 64,  64,  dc1, N1, k116, v116);
  vk_finalize<<<nb(N2 * 64,  T), T, 0, stream>>>(kv2f, 128, 0, 64,  64,  dc2, N2, k216, v216);

  // ---- attention branches (flash-style, WMMA, double-buffered LDS) ----
  dim3 agrid(L / (16 * ATT_WAVES), 4);  // 72 x 4 heads
  attn_wmma<<<agrid, 256, 0, stream>>>(qf16, 256, k016, v016, 128,
                                       oc16, 256, 0,   N0, 32, SCALE);
  attn_wmma<<<agrid, 256, 0, stream>>>(q12f16, 128, k116, v116, 64,
                                       oc16, 256, 128, N1, 16, SCALE);
  attn_wmma<<<agrid, 256, 0, stream>>>(q12f16 + 64, 128, k216, v216, 64,
                                       oc16, 256, 192, N2, 16, SCALE);

  // ---- output projection ----
  gemm_wmma_f16<<<nb((L / 16) * (256 / 32), 4), 128, 0, stream>>>(
      oc16, 256, wproj, 256, out, 256, L, 256, 256, proj_b);
}
